// BayesianGRU_21440476742259
// MI455X (gfx1250) — matrix operand order repaired and checked by reading, not compiled, not run on hardware
//
#include <hip/hip_runtime.h>

// Problem dims (from reference): B=64, T=256, IN=1024, HS=1024
#define BB   64
#define TT   256
#define INF  1024
#define HSZ  1024
#define G4   (4 * HSZ)

typedef __attribute__((ext_vector_type(16))) __bf16 v16bf;
typedef __attribute__((ext_vector_type(8)))  float  v8f;
typedef __attribute__((ext_vector_type(4)))  unsigned int u32x4;
typedef __attribute__((ext_vector_type(8)))  int i32x8;
typedef __attribute__((ext_vector_type(4)))  int i32x4;
typedef unsigned short bf16_t;

union FragBF {
    uint4 u[2];
    v16bf v;
};

// ---- float -> bf16 (round to nearest even) ----
__device__ __forceinline__ bf16_t f32_to_bf16(float f) {
    unsigned int u = __builtin_bit_cast(unsigned int, f);
    unsigned int lsb = (u >> 16) & 1u;
    u += 0x7fffu + lsb;
    return (bf16_t)(u >> 16);
}

__device__ __forceinline__ float softplusf(float x) {
    return (x > 20.f) ? x : log1pf(__expf(x));
}

// ============================================================================
// Kernel 1: sample w = mu + softplus(rho)*eps and TRANSPOSE [K,N] -> [N,K] bf16
// ============================================================================
__global__ void sample_transpose_kernel(const float* __restrict__ mu,
                                        const float* __restrict__ rho,
                                        const float* __restrict__ eps,
                                        bf16_t* __restrict__ wt,
                                        int K, int N) {
    __shared__ bf16_t tile[32][33];
    const int kb = blockIdx.x * 32;
    const int nb = blockIdx.y * 32;
    const int tx = threadIdx.x;   // 0..31
    const int ty = threadIdx.y;   // 0..7
    #pragma unroll
    for (int i = 0; i < 32; i += 8) {
        const int k = kb + ty + i;
        const int n = nb + tx;
        const size_t idx = (size_t)k * N + n;
        const float w = mu[idx] + softplusf(rho[idx]) * eps[idx];
        tile[ty + i][tx] = f32_to_bf16(w);
    }
    __syncthreads();
    #pragma unroll
    for (int i = 0; i < 32; i += 8) {
        const int n = nb + ty + i;
        const int k = kb + tx;
        wt[(size_t)n * K + k] = tile[tx][ty + i];
    }
}

__global__ void sample_bias_kernel(const float* __restrict__ mu,
                                   const float* __restrict__ rho,
                                   const float* __restrict__ eps,
                                   float* __restrict__ b, int n) {
    const int i = blockIdx.x * blockDim.x + threadIdx.x;
    if (i < n) b[i] = mu[i] + softplusf(rho[i]) * eps[i];
}

__global__ void cvt_bf16_kernel(const float* __restrict__ x,
                                bf16_t* __restrict__ xb, size_t n) {
    const size_t i = (size_t)blockIdx.x * blockDim.x + threadIdx.x;
    if (i < n) xb[i] = f32_to_bf16(x[i]);
}

__global__ void zero_h_kernel(float* __restrict__ hf, bf16_t* __restrict__ hb, int n) {
    const int i = blockIdx.x * blockDim.x + threadIdx.x;
    if (i < n) { hf[i] = 0.f; hb[i] = 0; }
}

// ============================================================================
// B fragment: 32x16 bf16 from transposed weights wt[N,K]; lane -> N = lane%16,
// lanes 0-15 hold K = k0+0..15, lanes 16-31 hold K = k0+16..31 (contiguous).
// ============================================================================
__device__ __forceinline__ v16bf load_b_frag(const bf16_t* __restrict__ row_base,
                                             int lane) {
    const int n    = lane & 15;
    const int koff = (lane >> 4) << 4;   // 0 or 16
    const bf16_t* p = row_base + (size_t)n * 1024 + koff;
    FragBF f;
    f.u[0] = *(const uint4*)(p);
    f.u[1] = *(const uint4*)(p + 8);
    return f.v;
}

// A fragment (16x32 bf16) from an LDS chunk laid out [64 rows][128 k] bf16.
// lane -> M = ms*16 + lane%16; halfs 0..15 -> K = ks + (lane<16?0:16)+i
// (contiguous, matching the B fragment).
__device__ __forceinline__ v16bf load_a_frag_lds(const bf16_t* abuf, int ms,
                                                 int ks, int lane) {
    const int m    = ms * 16 + (lane & 15);
    const int koff = (lane >> 4) << 4;   // 0 or 16
    const bf16_t* p = abuf + m * 128 + ks + koff;
    FragBF f;
    f.u[0] = *(const uint4*)(p);
    f.u[1] = *(const uint4*)(p + 8);
    return f.v;
}

// ============================================================================
// TDM: async-load a 64x128 bf16 tile (rows strided by `stride` elements in
// global memory) into LDS at byte offset lds_off.  D# per CDNA5 ISA 8.3/8.4:
//   group0: count=1, lds_addr, 57-bit global_addr, type=2
//   group1: data_size=2B, tensor_dim0=128, tensor_dim1=64, tile=128x64,
//           tensor_dim0_stride=stride
// 6-arg builtin variant (clang-23 / therock-10.0 headers).
// ============================================================================
__device__ __forceinline__ void tdm_load_tile(const bf16_t* gaddr,
                                              unsigned lds_off,
                                              unsigned long long stride) {
    const unsigned long long a = (unsigned long long)(uintptr_t)gaddr;
    u32x4 g0;
    g0.x = 1u;                                            // count=1, user desc
    g0.y = lds_off;                                       // LDS byte address
    g0.z = (unsigned)(a & 0xFFFFFFFFu);                   // global_addr[31:0]
    g0.w = ((unsigned)(a >> 32) & 0x01FFFFFFu) | 0x80000000u; // addr[56:32]|type=2
    i32x8 g1;
    g1[0] = 0x00010000;              // workgroup_mask=0, data_size=1 (2 bytes)
    g1[1] = (int)(128u << 16);       // tensor_dim0 = 128 (bits 79:48, low half)
    g1[2] = (int)(64u << 16);        // tensor_dim0 hi=0 | tensor_dim1 = 64
    g1[3] = (int)(128u << 16);       // tensor_dim1 hi=0 | tile_dim0 = 128
    g1[4] = 64;                      // tile_dim1 = 64, tile_dim2 = 0
    g1[5] = (int)(unsigned)(stride & 0xFFFFFFFFu);  // dim0_stride[31:0]
    g1[6] = (int)(unsigned)(stride >> 32);          // dim0_stride[47:32]
    g1[7] = 0;
    const i32x4 gz4 = {0, 0, 0, 0};            // groups 2/3 unused (2D tile)
    const i32x8 gz8 = {0, 0, 0, 0, 0, 0, 0, 0};
    __builtin_amdgcn_tensor_load_to_lds(g0, g1, gz4, gz4, gz8, 0);
}

// ============================================================================
// One GRU timestep.
//   grid = 64 blocks (16-col hidden strips), block = 128 threads = 4 waves.
//   Wave g computes gate g for ALL 64 batch rows (4 M-subtiles) => every
//   weight byte is read exactly once per step (16 MB/step, L2-resident).
//   A-operand (x_t rows then h rows, K=2048 total) is staged into LDS by the
//   Tensor Data Mover in 16 chunks of 64x128 bf16, double-buffered with
//   s_wait_tensorcnt; all 4 waves consume it via ds_load_b128 fragments.
// ============================================================================
__global__ __launch_bounds__(128) void gru_step_kernel(
    const bf16_t* __restrict__ xb,     // [B, T, IN] bf16
    const bf16_t* __restrict__ wihT,   // [4H, IN]  bf16 (transposed)
    const bf16_t* __restrict__ whhT,   // [4H, HS]  bf16 (transposed)
    const float*  __restrict__ bias,   // [4H]
    const float*  __restrict__ h_in,   // [B, HS] f32
    const bf16_t* __restrict__ h_in_b, // [B, HS] bf16
    float*  __restrict__ h_out,        // [B, HS] f32
    bf16_t* __restrict__ h_out_b,      // [B, HS] bf16
    float*  __restrict__ out,          // [B, T, HS] hidden_seq
    float*  __restrict__ h_last,       // [B, HS] or nullptr
    int t) {
    __shared__ __align__(16) bf16_t sA[2 * 64 * 128];   // 32 KB double buffer
    __shared__ float sG[4][64][16];                     // 16 KB gate exchange

    const int lane = threadIdx.x & 31;
    const int g    = threadIdx.x >> 5;      // wave id == gate id (0..3)
    const int n0   = blockIdx.x * 16;       // hidden column strip

    v8f acc[4] = {};                        // 4 M-subtiles (rows 0..63)

    const bf16_t* wg_ih = wihT + (size_t)(g * HSZ + n0) * INF;  // row stride 1024
    const bf16_t* wg_hh = whhT + (size_t)(g * HSZ + n0) * HSZ;

    // preload chunk 0 (x_t, k=[0,128))
    if (g == 0)
        tdm_load_tile(xb + (size_t)t * INF,
                      (unsigned)(uintptr_t)&sA[0],
                      (unsigned long long)TT * INF);

    for (int c = 0; c < 16; ++c) {
        if (g == 0) {
            const int cn = c + 1;
            if (cn < 16) {
                const bf16_t* base;
                unsigned long long stride;
                if (cn < 8) { base = xb + (size_t)t * INF + cn * 128;
                              stride = (unsigned long long)TT * INF; }
                else        { base = h_in_b + (cn - 8) * 128;
                              stride = HSZ; }
                tdm_load_tile(base, (unsigned)(uintptr_t)&sA[(cn & 1) * 8192],
                              stride);
                __builtin_amdgcn_s_wait_tensorcnt((short)1);
            } else {
                __builtin_amdgcn_s_wait_tensorcnt((short)0);
            }
        }
        __syncthreads();                    // chunk c ready for all waves

        const bf16_t* wrow = (c < 8) ? (wg_ih + c * 128) : (wg_hh + (c - 8) * 128);
        const bf16_t* abuf = &sA[(c & 1) * 8192];
        __builtin_prefetch((const void*)(wrow + 128), 0, 3);  // next K chunk
        #pragma unroll
        for (int ks = 0; ks < 128; ks += 32) {
            const v16bf b = load_b_frag(wrow + ks, lane);
            #pragma unroll
            for (int ms = 0; ms < 4; ++ms) {
                const v16bf a = load_a_frag_lds(abuf, ms, ks, lane);
                acc[ms] = __builtin_amdgcn_wmma_f32_16x16x32_bf16(
                    false, a, false, b, (short)0, acc[ms], false, false);
            }
        }
        __syncthreads();                    // done with buf before TDM reuses it
    }

    // ---- exchange gate tiles through LDS ----
    // C/D layout: VGPR i of acc[ms] -> M = ms*16 + i + 8*(lane>=16), N = lane%16
    {
        const int nl    = lane & 15;
        const int mbase = (lane >> 4) << 3;
        #pragma unroll
        for (int ms = 0; ms < 4; ++ms)
            #pragma unroll
            for (int i = 0; i < 8; ++i)
                sG[g][ms * 16 + mbase + i][nl] = acc[ms][i];
    }
    __syncthreads();

    // ---- gate math (f32) + stores: 64x16 elements over 128 threads ----
    #pragma unroll
    for (int e = threadIdx.x; e < 64 * 16; e += 128) {
        const int   m     = e >> 4;
        const int   nl    = e & 15;
        const int   n     = n0 + nl;
        const float hprev = h_in[(size_t)m * HSZ + n];
        const float z  = 1.f / (1.f + __expf(-(sG[0][m][nl] + bias[0 * HSZ + n])));
        const float r  = 1.f / (1.f + __expf(-(sG[1][m][nl] + bias[1 * HSZ + n])));
        const float nn = tanhf((sG[2][m][nl] + bias[2 * HSZ + n]) * r +
                               (sG[3][m][nl] + bias[3 * HSZ + n]));
        const float hn = (1.f - z) * nn + z * hprev;
        h_out[(size_t)m * HSZ + n]   = hn;
        h_out_b[(size_t)m * HSZ + n] = f32_to_bf16(hn);
        out[((size_t)m * TT + t) * HSZ + n] = hn;
        if (h_last) h_last[(size_t)m * HSZ + n] = hn;
    }
}

// ============================================================================
// Host side
// ============================================================================
extern "C" void kernel_launch(void* const* d_in, const int* in_sizes, int n_in,
                              void* d_out, int out_size, void* d_ws, size_t ws_size,
                              hipStream_t stream) {
    (void)in_sizes; (void)n_in; (void)out_size; (void)ws_size;

    const float* x       = (const float*)d_in[0];
    const float* wih_mu  = (const float*)d_in[1];
    const float* wih_rho = (const float*)d_in[2];
    const float* eps_ih  = (const float*)d_in[3];
    const float* whh_mu  = (const float*)d_in[4];
    const float* whh_rho = (const float*)d_in[5];
    const float* eps_hh  = (const float*)d_in[6];
    const float* b_mu    = (const float*)d_in[7];
    const float* b_rho   = (const float*)d_in[8];
    const float* eps_b   = (const float*)d_in[9];

    float* out    = (float*)d_out;                     // [B,T,HS] then [B,HS]
    float* h_last = out + (size_t)BB * TT * HSZ;

    uintptr_t p = (uintptr_t)d_ws;
    auto carve = [&](size_t bytes) -> void* {
        uintptr_t r = (p + 255) & ~(uintptr_t)255;
        p = r + bytes;
        return (void*)r;
    };
    bf16_t* wihT  = (bf16_t*)carve((size_t)G4 * INF * sizeof(bf16_t));
    bf16_t* whhT  = (bf16_t*)carve((size_t)G4 * HSZ * sizeof(bf16_t));
    float*  biasS = (float*) carve((size_t)G4 * sizeof(float));
    bf16_t* xb    = (bf16_t*)carve((size_t)BB * TT * INF * sizeof(bf16_t));
    float*  hA    = (float*) carve((size_t)BB * HSZ * sizeof(float));
    float*  hB    = (float*) carve((size_t)BB * HSZ * sizeof(float));
    bf16_t* hAb   = (bf16_t*)carve((size_t)BB * HSZ * sizeof(bf16_t));
    bf16_t* hBb   = (bf16_t*)carve((size_t)BB * HSZ * sizeof(bf16_t));

    sample_transpose_kernel<<<dim3(INF / 32, G4 / 32), dim3(32, 8), 0, stream>>>(
        wih_mu, wih_rho, eps_ih, wihT, INF, G4);
    sample_transpose_kernel<<<dim3(HSZ / 32, G4 / 32), dim3(32, 8), 0, stream>>>(
        whh_mu, whh_rho, eps_hh, whhT, HSZ, G4);
    sample_bias_kernel<<<G4 / 256, 256, 0, stream>>>(b_mu, b_rho, eps_b, biasS, G4);
    const size_t nx = (size_t)BB * TT * INF;
    cvt_bf16_kernel<<<(unsigned)((nx + 255) / 256), 256, 0, stream>>>(x, xb, nx);
    const int nh = BB * HSZ;
    zero_h_kernel<<<(nh + 255) / 256, 256, 0, stream>>>(hA, hAb, nh);

    float*  hin = hA;  float*  hout = hB;
    bf16_t* hib = hAb; bf16_t* hob  = hBb;
    for (int t = 0; t < TT; ++t) {
        gru_step_kernel<<<dim3(64), 128, 0, stream>>>(
            xb, wihT, whhT, biasS, hin, hib, hout, hob, out,
            (t == TT - 1) ? h_last : nullptr, t);
        float*  tf = hin; hin = hout; hout = tf;
        bf16_t* tb = hib; hib = hob;  hob  = tb;
    }
}
